// STAM_15247133901634
// MI455X (gfx1250) — matrix operand order repaired and checked by reading, not compiled, not run on hardware
//
#include <hip/hip_runtime.h>
#include <hip/hip_bf16.h>

// ---------------- problem constants ----------------
#define BB 1024
#define LL 200
#define DD 512
#define HH 8
#define DH 64
#define LP 224   // L padded to multiple of 32 (K-dim padding for WMMA)
#define LT 13    // number of 16-row tiles covering 200 (13*16 = 208)

typedef __attribute__((ext_vector_type(16))) __bf16          bf16x16;
typedef __attribute__((ext_vector_type(8)))  float           f32x8;
typedef __attribute__((ext_vector_type(4)))  float           f32x4;
typedef __attribute__((ext_vector_type(16))) unsigned short  u16x16;
typedef __attribute__((ext_vector_type(8)))  unsigned short  u16x8;
typedef __attribute__((ext_vector_type(4)))  unsigned short  u16x4;

__device__ __forceinline__ unsigned short f2bf(float f) {
  unsigned int u = __builtin_bit_cast(unsigned int, f);
  u += 0x7FFFu + ((u >> 16) & 1u);            // round-to-nearest-even
  return (unsigned short)(u >> 16);
}

union AFrag { u16x8 h[2]; u16x16 v; };

// A operand (16x32, row-major source): lane (g, m): halves0..7 = row[kb+8g..+7],
// halves8..15 = row[kb+16+8g..+7]   (two 16B loads)
__device__ __forceinline__ bf16x16 load_a_frag(const unsigned short* row, int kb, int g) {
  AFrag u;
  u.h[0] = *(const u16x8*)(row + kb + 8 * g);
  u.h[1] = *(const u16x8*)(row + kb + 16 + 8 * g);
  return __builtin_bit_cast(bf16x16, u.v);
}
// B operand (32x16, source stored N-major [N][K]): lane (g, n): halves0..7 = rowN[kb+8g..+7],
// halves8..15 = rowN[kb+16+8g..+7]   (two 16B loads)
__device__ __forceinline__ bf16x16 load_b_frag(const unsigned short* rowN, int kb, int g) {
  AFrag u;
  u.h[0] = *(const u16x8*)(rowN + kb + 8 * g);
  u.h[1] = *(const u16x8*)(rowN + kb + 16 + 8 * g);
  return __builtin_bit_cast(bf16x16, u.v);
}
__device__ __forceinline__ f32x8 wmma_bf16(bf16x16 a, bf16x16 b, f32x8 c) {
  return __builtin_amdgcn_wmma_f32_16x16x32_bf16(false, a, false, b, (short)0, c, false, false);
}

// ---------------- K1: x = inputs + pos_emb (f32 + bf16 copies) ----------------
__global__ __launch_bounds__(256) void k_prep_x(const float* __restrict__ in,
                                                const float* __restrict__ pos,
                                                float* __restrict__ xf,
                                                unsigned short* __restrict__ xb) {
  long i4 = (long)blockIdx.x * blockDim.x + threadIdx.x;
  long e = i4 * 4;
  if (e >= (long)BB * LL * DD) return;
  int d = (int)(e % DD);
  int l = (int)((e / DD) % LL);
  f32x4 a = *(const f32x4*)(in + e);
  f32x4 p = *(const f32x4*)(pos + (long)l * DD + d);
  f32x4 r = a + p;
  *(f32x4*)(xf + e) = r;
  u16x4 bb;
  bb.x = f2bf(r.x); bb.y = f2bf(r.y); bb.z = f2bf(r.z); bb.w = f2bf(r.w);
  *(u16x4*)(xb + e) = bb;
}

// ---------------- K2a: W [K][N] f32 -> WT [N][K] bf16 ----------------
__global__ __launch_bounds__(256) void k_transpose_w(const float* __restrict__ W,
                                                     unsigned short* __restrict__ WT) {
  int o = blockIdx.x * 256 + threadIdx.x;
  if (o >= DD * DD) return;
  int nrow = o / DD, kk = o % DD;
  WT[o] = f2bf(W[kk * DD + nrow]);
}

// ---------------- K2b: conv_w [200][200] f32 -> [200][224] bf16, zero-padded cols ----
__global__ __launch_bounds__(256) void k_pad_convw(const float* __restrict__ cw,
                                                   unsigned short* __restrict__ cwp) {
  int o = blockIdx.x * 256 + threadIdx.x;
  if (o >= LL * LP) return;
  int l = o / LP, m = o % LP;
  cwp[o] = (m < LL) ? f2bf(cw[l * LL + m]) : (unsigned short)0;
}

// ------- K3: QKV GEMM; one block per (b, mtile), x-tile staged once in LDS, ----------
// ------- all three projections from it; B-fragments double-buffered so each ---------
// ------- WMMA quad runs behind the next quad's loads (x read 1x from HBM). ----------
__global__ __launch_bounds__(256) void k_qkv(const unsigned short* __restrict__ xb,
                                             const unsigned short* __restrict__ WT0,
                                             const unsigned short* __restrict__ WT1,
                                             const unsigned short* __restrict__ WT2,
                                             unsigned short* __restrict__ qo,
                                             unsigned short* __restrict__ ko,
                                             unsigned short* __restrict__ vo) {
  const int b = blockIdx.x, mt = blockIdx.y;
  __shared__ unsigned short xs[16][DD];        // 16 KB A-tile
  const int tid = threadIdx.x;
  const int lane = tid & 31, wave = tid >> 5;
  const int g = lane >> 4, ln = lane & 15;

  // stage A tile (rows clamped) with 16B vector copies: 1024 chunks of 8
  for (int c = tid; c < 16 * DD / 8; c += 256) {
    int row = c >> 6;                 // 64 chunks per row
    int col = (c & 63) * 8;
    int l = min(mt * 16 + row, LL - 1);
    *(u16x8*)&xs[row][col] = *(const u16x8*)(xb + ((long)b * LL + l) * DD + col);
  }
  __syncthreads();

  const int nt0 = wave * 4;                 // 8 waves x 4 N-tiles = 32 tiles = 512 cols
  const unsigned short* arow = &xs[ln][0];
  for (int z = 0; z < 3; ++z) {
    const unsigned short* WT = (z == 0) ? WT0 : ((z == 1) ? WT1 : WT2);
    unsigned short* out = (z == 0) ? qo : ((z == 1) ? ko : vo);
    const unsigned short* brow[4];
#pragma unroll
    for (int i = 0; i < 4; ++i) brow[i] = WT + (long)((nt0 + i) * 16 + ln) * DD;

    f32x8 acc[4] = {};
    bf16x16 bfr[2][4];
#pragma unroll
    for (int i = 0; i < 4; ++i) bfr[0][i] = load_b_frag(brow[i], 0, g);

#pragma unroll
    for (int s = 0; s < DD / 32; ++s) {       // 16 k-steps
      const int p = s & 1;
      if (s + 1 < DD / 32) {
#pragma unroll
        for (int i = 0; i < 4; ++i) bfr[p ^ 1][i] = load_b_frag(brow[i], (s + 1) * 32, g);
      }
      bf16x16 a = load_a_frag(arow, s * 32, g);
#pragma unroll
      for (int i = 0; i < 4; ++i) acc[i] = wmma_bf16(a, bfr[p][i], acc[i]);
    }
#pragma unroll
    for (int i = 0; i < 4; ++i) {
      int ncol = (nt0 + i) * 16 + ln;
#pragma unroll
      for (int r = 0; r < 8; ++r) {
        int l = mt * 16 + r + 8 * g;
        if (l < LL) out[((long)b * LL + l) * DD + ncol] = f2bf(acc[i][r]);
      }
    }
  }
}

// ------- K4: fused attention; one block per (b, head); k/vT staged once in ----------
// ------- dynamic LDS (CDNA5 WGP LDS > 64KB), loop over all 13 l-tiles. --------------
#define SM_S   0
#define SM_QS  (SM_S + 16 * LP * 4)                 // 14336
#define SM_KS  (SM_QS + 16 * DH * 2)                // +2048
#define SM_VT  (SM_KS + LP * DH * 2)                // +28672
#define SM_ATT (SM_VT + DH * LP * 2)                // +28672
#define SM_TOT (SM_ATT + 16 * LP * 2)               // +7168 = 80896 bytes

__global__ __launch_bounds__(256) void k_attn(const unsigned short* __restrict__ q,
                                              const unsigned short* __restrict__ k,
                                              const unsigned short* __restrict__ v,
                                              float* __restrict__ hf,
                                              unsigned short* __restrict__ hT) {
  extern __shared__ char smem[];
  float*          S   = (float*)(smem + SM_S);            // [16][LP]
  unsigned short* qs  = (unsigned short*)(smem + SM_QS);  // [16][DH]
  unsigned short* ks  = (unsigned short*)(smem + SM_KS);  // [LP][DH] row-major keys
  unsigned short* vT  = (unsigned short*)(smem + SM_VT);  // [DH][LP] transposed values
  unsigned short* att = (unsigned short*)(smem + SM_ATT); // [16][LP]

  const int b = blockIdx.x, head = blockIdx.y;
  const int tid = threadIdx.x;
  const int lane = tid & 31, wave = tid >> 5;
  const int g = lane >> 4, ln = lane & 15;

  // stage keys (row-major, zero-padded) and values (transposed, zero-padded) once
  for (int idx = tid; idx < LP * DH; idx += 256) {
    int m = idx / DH, d = idx % DH;
    unsigned short kv = 0, vv = 0;
    if (m < LL) {
      long base = ((long)b * LL + m) * DD + head * DH + d;
      kv = k[base];
      vv = v[base];
    }
    ks[m * DH + d] = kv;
    vT[d * LP + m] = vv;
  }

  for (int lt = 0; lt < LT; ++lt) {
    // stage q tile (16 x 64)
    for (int idx = tid; idx < 16 * DH; idx += 256) {
      int r = idx / DH, d = idx % DH;
      int l = min(lt * 16 + r, LL - 1);
      qs[idx] = q[((long)b * LL + l) * DD + head * DH + d];
    }
    __syncthreads();

    // scores: S[16 x 224] = q_tile @ k^T / sqrt(200)
    {
      const unsigned short* arow = qs + ln * DH;
      bf16x16 a0 = load_a_frag(arow, 0, g);
      bf16x16 a1 = load_a_frag(arow, 32, g);
      for (int nt = wave; nt < 14; nt += 8) {
        const unsigned short* brow = ks + (nt * 16 + ln) * DH;
        f32x8 acc = {};
        acc = wmma_bf16(a0, load_b_frag(brow, 0, g), acc);
        acc = wmma_bf16(a1, load_b_frag(brow, 32, g), acc);
#pragma unroll
        for (int r = 0; r < 8; ++r)
          S[(r + 8 * g) * LP + nt * 16 + ln] = acc[r] * 0.07071067811865475f;
      }
    }
    __syncthreads();

    // softmax over m<200 (each wave: 2 rows; 7 cols per lane)
#pragma unroll
    for (int rr = 0; rr < 2; ++rr) {
      int row = wave * 2 + rr;
      float vals[7];
      float mx = -1e30f;
#pragma unroll
      for (int j = 0; j < 7; ++j) {
        int m = lane + j * 32;
        float s = (m < LL) ? S[row * LP + m] : -1e30f;
        vals[j] = s;
        mx = fmaxf(mx, s);
      }
      for (int off = 16; off; off >>= 1) mx = fmaxf(mx, __shfl_xor(mx, off, 32));
      float sum = 0.f;
#pragma unroll
      for (int j = 0; j < 7; ++j) {
        int m = lane + j * 32;
        float e = (m < LL) ? __expf(vals[j] - mx) : 0.f;
        vals[j] = e;
        sum += e;
      }
      for (int off = 16; off; off >>= 1) sum += __shfl_xor(sum, off, 32);
      float inv = 1.f / sum;
#pragma unroll
      for (int j = 0; j < 7; ++j) att[row * LP + lane + j * 32] = f2bf(vals[j] * inv);
    }
    __syncthreads();

    // h_tile [16 x 64] = att @ v   (waves 0..3, one 16-wide d tile each, K=224)
    if (wave < 4) {
      const unsigned short* arow = att + ln * LP;
      const unsigned short* brow = vT + (wave * 16 + ln) * LP;
      f32x8 acc = {};
#pragma unroll
      for (int kk = 0; kk < LP; kk += 32)
        acc = wmma_bf16(load_a_frag(arow, kk, g), load_b_frag(brow, kk, g), acc);
      int dglob = head * DH + wave * 16 + ln;
#pragma unroll
      for (int r = 0; r < 8; ++r) {
        int l = lt * 16 + r + 8 * g;
        if (l < LL) {
          long idx = ((long)b * LL + l) * DD + dglob;
          hf[idx] = acc[r];
          hT[((long)b * DD + dglob) * LP + l] = f2bf(acc[r]);
        }
      }
    }
    __syncthreads();   // protect S (next scores) / att (next softmax) reuse
  }
}

// ------- K5: time-mix conv GEMM + residuals; one block per (b, 128-wide d block); ----
// ------- hT B-fragments hoisted to registers, all 13 l-tiles looped (hT read 1x). ----
__global__ __launch_bounds__(256) void k_conv(const unsigned short* __restrict__ cwp,
                                              const unsigned short* __restrict__ hT,
                                              const float* __restrict__ hf,
                                              const float* __restrict__ xf,
                                              const float* __restrict__ cb,
                                              float* __restrict__ out) {
  const int b = blockIdx.x, dblk = blockIdx.y;
  const int lane = threadIdx.x & 31, wave = threadIdx.x >> 5;
  const int g = lane >> 4, ln = lane & 15;
  const int dglob = dblk * 128 + wave * 16 + ln;
  const unsigned short* brow = hT + ((long)b * DD + dglob) * LP;
  bf16x16 bregs[7];
#pragma unroll
  for (int i = 0; i < 7; ++i) bregs[i] = load_b_frag(brow, i * 32, g);

  for (int lt = 0; lt < LT; ++lt) {
    const unsigned short* arow = cwp + (long)min(lt * 16 + ln, LL - 1) * LP;
    f32x8 acc = {};
#pragma unroll
    for (int i = 0; i < 7; ++i)
      acc = wmma_bf16(load_a_frag(arow, i * 32, g), bregs[i], acc);
#pragma unroll
    for (int r = 0; r < 8; ++r) {
      int l = lt * 16 + r + 8 * g;
      if (l < LL) {
        long idx = ((long)b * LL + l) * DD + dglob;
        out[idx] = acc[r] + xf[idx] + hf[idx] + cb[l];
      }
    }
  }
}

// ---------------- K6: in-place L2 normalize over D (one wave per row) ----------------
__global__ __launch_bounds__(256) void k_norm(float* __restrict__ out) {
  const int lane = threadIdx.x & 31, wave = threadIdx.x >> 5;
  long row = (long)blockIdx.x * 8 + wave;
  if (row >= (long)BB * LL) return;
  float* p = out + row * DD;
  float vals[16];
  float ss = 0.f;
#pragma unroll
  for (int j = 0; j < 16; ++j) {
    float x = p[lane + j * 32];
    vals[j] = x;
    ss += x * x;
  }
  for (int off = 16; off; off >>= 1) ss += __shfl_xor(ss, off, 32);
  float inv = 1.f / fmaxf(sqrtf(ss), 1e-12f);
#pragma unroll
  for (int j = 0; j < 16; ++j) p[lane + j * 32] = vals[j] * inv;
}

// ---------------- launch ----------------
extern "C" void kernel_launch(void* const* d_in, const int* in_sizes, int n_in,
                              void* d_out, int out_size, void* d_ws, size_t ws_size,
                              hipStream_t stream) {
  (void)in_sizes; (void)n_in; (void)out_size; (void)ws_size;
  const float* inputs = (const float*)d_in[0];
  const float* pos    = (const float*)d_in[1];
  const float* Wq     = (const float*)d_in[2];
  const float* Wk     = (const float*)d_in[3];
  const float* Wv     = (const float*)d_in[4];
  const float* cw     = (const float*)d_in[5];
  const float* cb     = (const float*)d_in[6];
  float* out = (float*)d_out;

  char* w = (char*)d_ws;
  auto alloc = [&](size_t bytes) {
    char* p = w;
    w += (bytes + 255) & ~(size_t)255;
    return p;
  };
  float*          xf  = (float*)         alloc((size_t)BB * LL * DD * 4);
  float*          hf  = (float*)         alloc((size_t)BB * LL * DD * 4);
  unsigned short* xb  = (unsigned short*)alloc((size_t)BB * LL * DD * 2);
  unsigned short* qb  = (unsigned short*)alloc((size_t)BB * LL * DD * 2);
  unsigned short* kb  = (unsigned short*)alloc((size_t)BB * LL * DD * 2);
  unsigned short* vb  = (unsigned short*)alloc((size_t)BB * LL * DD * 2);
  unsigned short* hT  = (unsigned short*)alloc((size_t)BB * DD * LP * 2);
  unsigned short* WqT = (unsigned short*)alloc((size_t)DD * DD * 2);
  unsigned short* WkT = (unsigned short*)alloc((size_t)DD * DD * 2);
  unsigned short* WvT = (unsigned short*)alloc((size_t)DD * DD * 2);
  unsigned short* cwp = (unsigned short*)alloc((size_t)LL * LP * 2);

  k_prep_x<<<(BB * LL * DD / 4 + 255) / 256, 256, 0, stream>>>(inputs, pos, xf, xb);
  k_transpose_w<<<(DD * DD + 255) / 256, 256, 0, stream>>>(Wq, WqT);
  k_transpose_w<<<(DD * DD + 255) / 256, 256, 0, stream>>>(Wk, WkT);
  k_transpose_w<<<(DD * DD + 255) / 256, 256, 0, stream>>>(Wv, WvT);
  k_pad_convw<<<(LL * LP + 255) / 256, 256, 0, stream>>>(cw, cwp);
  k_qkv<<<dim3(BB, LT), 256, 0, stream>>>(xb, WqT, WkT, WvT, qb, kb, vb);
  k_attn<<<dim3(BB, HH), 256, SM_TOT, stream>>>(qb, kb, vb, hf, hT);
  k_conv<<<dim3(BB, 4), 256, 0, stream>>>(cwp, hT, hf, xf, cb, out);
  k_norm<<<(BB * LL + 7) / 8, 256, 0, stream>>>(out);
}
